// MambaLayer_34093450396622
// MI455X (gfx1250) — matrix operand order repaired and checked by reading, not compiled, not run on hardware
//
#include <hip/hip_runtime.h>
#include <hip/hip_bf16.h>
#include <math.h>

typedef __attribute__((ext_vector_type(16))) _Float16 v16h;
typedef __attribute__((ext_vector_type(8)))  float    v8f;

#define DIMC   384
#define LSEQ   4096
#define NB     4
#define MTOT   (NB * LSEQ)      // 16384 rows
#define DIN    768
#define DST    16
#define DTRANK 24
#define DTPAD  32               // dt K padded 24 -> 32
#define XPN    64               // x_proj N padded 56 -> 64
#define TC     64               // scan chunk length
#define NCH    (LSEQ / TC)      // 64 chunks per sequence

// ---------------------------------------------------------------------------
// Core WMMA tile: rows [m0,m0+32) of A (M x K f16, row-major) times
// cols [n0,n0+64) of W (N x K f16, row-major torch (out,in) layout):
// C[m][n] = sum_k A[m][k] * W[n][k].
// acc[g*4+j] covers rows m0+16g.., cols n0+16j..  (two A frags share each B frag)
// Per-lane operand layouts per CDNA5 ISA 7.12.2 (wave32).
// ---------------------------------------------------------------------------
__device__ __forceinline__ void wmma_tile_32x64(const _Float16* __restrict__ A,
                                                const _Float16* __restrict__ W,
                                                int K, int m0, int n0,
                                                v8f acc[8]) {
    const int lane  = threadIdx.x & 31;
    const int half  = lane >> 4;     // 0: lanes 0-15, 1: lanes 16-31
    const int idx16 = lane & 15;     // A row within group / B col within tile
    const _Float16* arow0 = A + (size_t)(m0 + idx16) * K;
    const _Float16* arow1 = arow0 + (size_t)16 * K;
    for (int k0 = 0; k0 < K; k0 += 32) {
        // A fragment: e<8 -> K = k0 + 8*half + e ; e>=8 -> K = k0+16+8*half+(e-8)
        v16h a0, a1;
        const _Float16* ap0 = arow0 + k0 + 8 * half;
        const _Float16* ap1 = arow1 + k0 + 8 * half;
#pragma unroll
        for (int e = 0; e < 8; ++e) {
            a0[e] = ap0[e];  a0[8 + e] = ap0[16 + e];
            a1[e] = ap1[e];  a1[8 + e] = ap1[16 + e];
        }
#pragma unroll
        for (int j = 0; j < 4; ++j) {
            // B fragment: e<8 -> K = k0 + 8*half + e ; e>=8 -> K = k0+16+8*half+(e-8), N = n0+16j+idx16
            v16h bf;
            const _Float16* wp = W + (size_t)(n0 + j * 16 + idx16) * K + k0 + 8 * half;
#pragma unroll
            for (int e = 0; e < 8; ++e) { bf[e] = wp[e]; bf[8 + e] = wp[16 + e]; }
            acc[j]     = __builtin_amdgcn_wmma_f32_16x16x32_f16(
                             false, a0, false, bf, (short)0, acc[j],     false, false);
            acc[4 + j] = __builtin_amdgcn_wmma_f32_16x16x32_f16(
                             false, a1, false, bf, (short)0, acc[4 + j], false, false);
        }
    }
}

// ---------------------------------------------------------------------------
// Weight convert + pad: out[n*Kp+k] = (n<N && k<K) ? f16(in[n*K+k]) : 0
// ---------------------------------------------------------------------------
__global__ void cvt_pad_kernel(const float* __restrict__ in, _Float16* __restrict__ out,
                               int N, int K, int Np, int Kp) {
    int idx = blockIdx.x * 256 + threadIdx.x;
    int tot = Np * Kp;
    if (idx >= tot) return;
    int n = idx / Kp, k = idx - n * Kp;
    float v = (n < N && k < K) ? in[(size_t)n * K + k] : 0.f;
    out[idx] = (_Float16)v;
}

// ---------------------------------------------------------------------------
// LayerNorm over channels; input x is (B, C, L), output xn f16 (B*L, C)
// ---------------------------------------------------------------------------
__global__ void ln_kernel(const float* __restrict__ x, const float* __restrict__ gamma,
                          const float* __restrict__ beta, _Float16* __restrict__ xn) {
    const int bl = blockIdx.x;            // b*L + l
    const int b = bl / LSEQ, l = bl - b * LSEQ;
    const float* xb = x + (size_t)b * DIMC * LSEQ + l;   // stride L over c
    __shared__ float s_sum[128], s_sq[128];
    const int t = threadIdx.x;
    float v[3], sum = 0.f, sq = 0.f;
#pragma unroll
    for (int i = 0; i < 3; ++i) {
        int c = t + i * 128;
        float val = xb[(size_t)c * LSEQ];
        v[i] = val; sum += val; sq += val * val;
    }
    s_sum[t] = sum; s_sq[t] = sq;
    __syncthreads();
    for (int off = 64; off > 0; off >>= 1) {
        if (t < off) { s_sum[t] += s_sum[t + off]; s_sq[t] += s_sq[t + off]; }
        __syncthreads();
    }
    const float mu  = s_sum[0] * (1.f / DIMC);
    const float var = s_sq[0] * (1.f / DIMC) - mu * mu;
    const float inv = rsqrtf(var + 1e-5f);
    _Float16* o = xn + (size_t)bl * DIMC;
#pragma unroll
    for (int i = 0; i < 3; ++i) {
        int c = t + i * 128;
        o[c] = (_Float16)((v[i] - mu) * inv * gamma[c] + beta[c]);
    }
}

// ---------------------------------------------------------------------------
// in_proj: (M x 384) @ (384 x 1536) -> xm (cols 0..767) / z (cols 768..1535)
// ---------------------------------------------------------------------------
__global__ void gemm_inproj(const _Float16* __restrict__ A, const _Float16* __restrict__ W,
                            _Float16* __restrict__ xm, _Float16* __restrict__ zh) {
    const int m0 = blockIdx.x * 32;
    const int n0 = blockIdx.y * 64;
    v8f acc[8] = {v8f{}, v8f{}, v8f{}, v8f{}, v8f{}, v8f{}, v8f{}, v8f{}};
    wmma_tile_32x64(A, W, DIMC, m0, n0, acc);
    const int lane = threadIdx.x & 31, half = lane >> 4, nr = lane & 15;
    const bool isZ = (n0 >= DIN);
    _Float16* dst = isZ ? zh : xm;
    const int nb = isZ ? (n0 - DIN) : n0;
#pragma unroll
    for (int g = 0; g < 2; ++g)
#pragma unroll
        for (int j = 0; j < 4; ++j)
#pragma unroll
            for (int r = 0; r < 8; ++r) {
                int m = m0 + g * 16 + r + 8 * half;
                int n = nb + j * 16 + nr;
                dst[(size_t)m * DIN + n] = (_Float16)acc[g * 4 + j][r];
            }
}

// ---------------------------------------------------------------------------
// Causal depthwise conv1d (k=4) + SiLU.  xm (B*L, 768) f16 -> xc f16
// ---------------------------------------------------------------------------
__global__ void conv_silu_kernel(const _Float16* __restrict__ xm,
                                 const float* __restrict__ cw, const float* __restrict__ cb,
                                 _Float16* __restrict__ xc) {
    size_t idx = (size_t)blockIdx.x * 256 + threadIdx.x;
    const size_t total = (size_t)MTOT * DIN;
    if (idx >= total) return;
    const int d  = (int)(idx % DIN);
    const size_t bl = idx / DIN;
    const int t = (int)(bl % LSEQ);
    float acc = cb[d];
#pragma unroll
    for (int j = 0; j < 4; ++j) {
        int tt = t - 3 + j;
        if (tt >= 0) acc += cw[d * 4 + j] * (float)xm[idx - (size_t)(3 - j) * DIN];
    }
    float s = acc / (1.f + __expf(-acc));
    xc[idx] = (_Float16)s;
}

// ---------------------------------------------------------------------------
// x_proj: (M x 768) @ (768 x 64pad) -> dtBC f32 (M x 64) + dt f16 (M x 32pad)
// cols: 0..23 dt, 24..39 B, 40..55 C, 56..63 pad
// ---------------------------------------------------------------------------
__global__ void gemm_xproj(const _Float16* __restrict__ A, const _Float16* __restrict__ W,
                           float* __restrict__ dtBC, _Float16* __restrict__ dth) {
    const int m0 = blockIdx.x * 32;
    v8f acc[8] = {v8f{}, v8f{}, v8f{}, v8f{}, v8f{}, v8f{}, v8f{}, v8f{}};
    wmma_tile_32x64(A, W, DIN, m0, 0, acc);
    const int lane = threadIdx.x & 31, half = lane >> 4, nr = lane & 15;
#pragma unroll
    for (int g = 0; g < 2; ++g)
#pragma unroll
        for (int j = 0; j < 4; ++j)
#pragma unroll
            for (int r = 0; r < 8; ++r) {
                int m = m0 + g * 16 + r + 8 * half;
                int n = j * 16 + nr;
                float v = acc[g * 4 + j][r];
                dtBC[(size_t)m * XPN + n] = v;
                if (n < DTPAD) dth[(size_t)m * DTPAD + n] = (_Float16)(n < DTRANK ? v : 0.f);
            }
}

// ---------------------------------------------------------------------------
// dt_proj: (M x 32pad) @ (32pad x 768) + b_dt -> softplus -> delta f32
// ---------------------------------------------------------------------------
__global__ void gemm_dtproj(const _Float16* __restrict__ A, const _Float16* __restrict__ W,
                            const float* __restrict__ bdt, float* __restrict__ delta) {
    const int m0 = blockIdx.x * 32;
    const int n0 = blockIdx.y * 64;
    v8f acc[8] = {v8f{}, v8f{}, v8f{}, v8f{}, v8f{}, v8f{}, v8f{}, v8f{}};
    wmma_tile_32x64(A, W, DTPAD, m0, n0, acc);
    const int lane = threadIdx.x & 31, half = lane >> 4, nr = lane & 15;
#pragma unroll
    for (int g = 0; g < 2; ++g)
#pragma unroll
        for (int j = 0; j < 4; ++j)
#pragma unroll
            for (int r = 0; r < 8; ++r) {
                int m = m0 + g * 16 + r + 8 * half;
                int n = n0 + j * 16 + nr;
                float v = acc[g * 4 + j][r] + bdt[n];
                float sp = (v > 20.f) ? v : log1pf(__expf(v));
                delta[(size_t)m * DIN + n] = sp;
            }
}

// ---------------------------------------------------------------------------
// Chunked selective scan.  Recurrence per (b,d,s):
//   h_t = e_t * h_{t-1} + u_t,  e_t = exp(delta_t * a_s),  u_t = delta_t*x_t*B_t[s]
// Chunk composition: S_c = P_c * S_{c-1} + H_c  with  P_c = prod e, H_c = zero-init scan.
// Pass 1 (parallel over chunks): compute P_c, H_c.
// Pass 2 (per channel, NCH steps): carry-in state per chunk.
// Pass 3 (parallel over chunks): rescan from carry, y = sum_s h*C, fuse D-skip + gate.
// One wave = 2 channels x 16 states.
// ---------------------------------------------------------------------------
__global__ void scan_chunk_kernel(const float* __restrict__ delta,
                                  const _Float16* __restrict__ xc,
                                  const float* __restrict__ dtBC,
                                  const float* __restrict__ A_log,
                                  float* __restrict__ Pbuf, float* __restrict__ Hbuf) {
    const int blk = blockIdx.x;                 // ((b*(DIN/2)+dp)*NCH + c)
    const int c   = blk % NCH;
    const int tmp = blk / NCH;
    const int dp  = tmp % (DIN / 2), b = tmp / (DIN / 2);
    const int lane = threadIdx.x & 31;
    const int s = lane & 15, sub = lane >> 4;
    const int d = dp * 2 + sub;
    const float a = -__expf(A_log[d * DST + s]);
    float P = 1.f, H = 0.f;
    const size_t rowD  = (size_t)b * LSEQ * DIN;
    const size_t rowBC = (size_t)b * LSEQ * XPN;
    const int t0 = c * TC;
    for (int i = 0; i < TC; ++i) {
        const int t = t0 + i;
        const size_t od = rowD + (size_t)t * DIN + d;
        float dlt = delta[od];
        float xv  = (float)xc[od];
        float Bv  = dtBC[rowBC + (size_t)t * XPN + 24 + s];
        float e   = __expf(dlt * a);
        P *= e;
        H = H * e + dlt * xv * Bv;
    }
    const size_t oidx = (((size_t)b * DIN + d) * NCH + c) * DST + s;
    Pbuf[oidx] = P;
    Hbuf[oidx] = H;
}

__global__ void scan_combine_kernel(const float* __restrict__ Pbuf,
                                    const float* __restrict__ Hbuf,
                                    float* __restrict__ carry) {
    const int blk = blockIdx.x;                 // b*(DIN/2)+dp
    const int dp = blk % (DIN / 2), b = blk / (DIN / 2);
    const int lane = threadIdx.x & 31;
    const int s = lane & 15, sub = lane >> 4;
    const int d = dp * 2 + sub;
    const size_t base = ((size_t)b * DIN + d) * NCH * DST + s;
    float S = 0.f;
    for (int c = 0; c < NCH; ++c) {
        const size_t idx = base + (size_t)c * DST;
        carry[idx] = S;                          // state entering chunk c
        S = Pbuf[idx] * S + Hbuf[idx];
    }
}

__global__ void scan_final_kernel(const float* __restrict__ delta,
                                  const _Float16* __restrict__ xc,
                                  const float* __restrict__ dtBC,
                                  const _Float16* __restrict__ zh,
                                  const float* __restrict__ A_log,
                                  const float* __restrict__ Dp,
                                  const float* __restrict__ carry,
                                  _Float16* __restrict__ yg) {
    const int blk = blockIdx.x;                 // ((b*(DIN/2)+dp)*NCH + c)
    const int c   = blk % NCH;
    const int tmp = blk / NCH;
    const int dp  = tmp % (DIN / 2), b = tmp / (DIN / 2);
    const int lane = threadIdx.x & 31;
    const int s = lane & 15, sub = lane >> 4;
    const int d = dp * 2 + sub;
    const float a  = -__expf(A_log[d * DST + s]);
    const float Dv = Dp[d];
    float h = carry[(((size_t)b * DIN + d) * NCH + c) * DST + s];
    const size_t rowD  = (size_t)b * LSEQ * DIN;
    const size_t rowBC = (size_t)b * LSEQ * XPN;
    const int t0 = c * TC;
    for (int i = 0; i < TC; ++i) {
        const int t = t0 + i;
        const size_t od  = rowD + (size_t)t * DIN + d;
        const size_t obc = rowBC + (size_t)t * XPN;
        float dlt = delta[od];
        float xv  = (float)xc[od];
        float Bv  = dtBC[obc + 24 + s];
        float Cv  = dtBC[obc + 40 + s];
        h = h * __expf(dlt * a) + dlt * xv * Bv;
        float p = h * Cv;
        p += __shfl_xor(p, 1, 32);
        p += __shfl_xor(p, 2, 32);
        p += __shfl_xor(p, 4, 32);
        p += __shfl_xor(p, 8, 32);
        if (s == 0) {
            float zv = (float)zh[od];
            float g  = zv / (1.f + __expf(-zv));
            yg[od] = (_Float16)((p + xv * Dv) * g);
        }
    }
}

// ---------------------------------------------------------------------------
// out_proj: (M x 768) @ (768 x 384) -> d_out with (B,L,C)->(B,C,L) transpose
// ---------------------------------------------------------------------------
__global__ void gemm_outproj(const _Float16* __restrict__ A, const _Float16* __restrict__ W,
                             float* __restrict__ out) {
    const int m0 = blockIdx.x * 32;
    const int n0 = blockIdx.y * 64;
    v8f acc[8] = {v8f{}, v8f{}, v8f{}, v8f{}, v8f{}, v8f{}, v8f{}, v8f{}};
    wmma_tile_32x64(A, W, DIN, m0, n0, acc);
    const int lane = threadIdx.x & 31, half = lane >> 4, nr = lane & 15;
#pragma unroll
    for (int g = 0; g < 2; ++g)
#pragma unroll
        for (int j = 0; j < 4; ++j)
#pragma unroll
            for (int r = 0; r < 8; ++r) {
                int m = m0 + g * 16 + r + 8 * half;   // b*L + l
                int b = m / LSEQ, l = m - b * LSEQ;
                int cc = n0 + j * 16 + nr;
                out[(size_t)b * DIMC * LSEQ + (size_t)cc * LSEQ + l] = acc[g * 4 + j][r];
            }
}

// ---------------------------------------------------------------------------
extern "C" void kernel_launch(void* const* d_in, const int* in_sizes, int n_in,
                              void* d_out, int out_size, void* d_ws, size_t ws_size,
                              hipStream_t stream) {
    const float* x      = (const float*)d_in[0];
    const float* gamma  = (const float*)d_in[1];
    const float* beta   = (const float*)d_in[2];
    const float* W_in   = (const float*)d_in[3];   // (1536, 384)
    const float* conv_w = (const float*)d_in[4];   // (768, 1, 4)
    const float* conv_b = (const float*)d_in[5];
    const float* W_xp   = (const float*)d_in[6];   // (56, 768)
    const float* W_dt   = (const float*)d_in[7];   // (768, 24)
    const float* b_dt   = (const float*)d_in[8];
    const float* A_log  = (const float*)d_in[9];   // (768, 16)
    const float* Dpar   = (const float*)d_in[10];
    const float* W_out  = (const float*)d_in[11];  // (384, 768)
    float* out = (float*)d_out;

    // ---- workspace layout (256-byte aligned regions) ----
    char* ws = (char*)d_ws;
    size_t o = 0;
    auto alloc = [&](size_t bytes) { size_t r = o; o += (bytes + 255) & ~(size_t)255; return r; };
    const size_t o_Win  = alloc((size_t)2 * DIN * DIMC * 2);        // f16 1536x384
    const size_t o_Wxp  = alloc((size_t)XPN * DIN * 2);             // f16 64x768 (pad)
    const size_t o_Wdt  = alloc((size_t)DIN * DTPAD * 2);           // f16 768x32 (pad)
    const size_t o_Wout = alloc((size_t)DIMC * DIN * 2);            // f16 384x768
    const size_t o_xn   = alloc((size_t)MTOT * DIMC * 2);           // f16; reused for dtBC+dt
    const size_t o_xm   = alloc((size_t)MTOT * DIN * 2);            // f16; reused for yg
    const size_t o_z    = alloc((size_t)MTOT * DIN * 2);            // f16
    const size_t o_xc   = alloc((size_t)MTOT * DIN * 2);            // f16
    const size_t o_dlt  = alloc((size_t)MTOT * DIN * 4);            // f32
    const size_t scanN  = (size_t)NB * DIN * NCH * DST * 4;         // 12.58 MB each
    const size_t o_P    = alloc(scanN);
    const size_t o_H    = alloc(scanN);
    const size_t o_cry  = alloc(scanN);
    // aliases (xn dead after in_proj; xm dead after conv)
    const size_t o_dtBC = o_xn;                                     // f32 M x 64
    const size_t o_dth  = o_xn + (size_t)MTOT * XPN * 4 + 256;      // f16 M x 32
    const size_t o_yg   = o_xm;                                     // f16 M x 768
    (void)ws_size; (void)in_sizes; (void)n_in; (void)out_size;

    _Float16* Wh_in  = (_Float16*)(ws + o_Win);
    _Float16* Wh_xp  = (_Float16*)(ws + o_Wxp);
    _Float16* Wh_dt  = (_Float16*)(ws + o_Wdt);
    _Float16* Wh_out = (_Float16*)(ws + o_Wout);
    _Float16* xn_h   = (_Float16*)(ws + o_xn);
    _Float16* xm_h   = (_Float16*)(ws + o_xm);
    _Float16* z_h    = (_Float16*)(ws + o_z);
    _Float16* xc_h   = (_Float16*)(ws + o_xc);
    float*    deltaf = (float*)   (ws + o_dlt);
    float*    Pbuf   = (float*)   (ws + o_P);
    float*    Hbuf   = (float*)   (ws + o_H);
    float*    carry  = (float*)   (ws + o_cry);
    float*    dtBC   = (float*)   (ws + o_dtBC);
    _Float16* dt_h   = (_Float16*)(ws + o_dth);
    _Float16* yg_h   = (_Float16*)(ws + o_yg);

    // 1) convert/pad weights to f16
    cvt_pad_kernel<<<(2 * DIN * DIMC + 255) / 256, 256, 0, stream>>>(W_in,  Wh_in,  2 * DIN, DIMC, 2 * DIN, DIMC);
    cvt_pad_kernel<<<(XPN * DIN + 255) / 256, 256, 0, stream>>>(W_xp,  Wh_xp,  56, DIN, XPN, DIN);
    cvt_pad_kernel<<<(DIN * DTPAD + 255) / 256, 256, 0, stream>>>(W_dt,  Wh_dt,  DIN, DTRANK, DIN, DTPAD);
    cvt_pad_kernel<<<(DIMC * DIN + 255) / 256, 256, 0, stream>>>(W_out, Wh_out, DIMC, DIN, DIMC, DIN);

    // 2) layernorm -> xn f16
    ln_kernel<<<MTOT, 128, 0, stream>>>(x, gamma, beta, xn_h);

    // 3) in_proj (WMMA) -> xm, z
    gemm_inproj<<<dim3(MTOT / 32, (2 * DIN) / 64), 32, 0, stream>>>(xn_h, Wh_in, xm_h, z_h);

    // 4) causal depthwise conv + SiLU -> xc
    conv_silu_kernel<<<(int)(((size_t)MTOT * DIN + 255) / 256), 256, 0, stream>>>(xm_h, conv_w, conv_b, xc_h);

    // 5) x_proj (WMMA) -> dtBC f32 + dt f16
    gemm_xproj<<<dim3(MTOT / 32, 1), 32, 0, stream>>>(xc_h, Wh_xp, dtBC, dt_h);

    // 6) dt_proj (WMMA) + bias + softplus -> delta f32
    gemm_dtproj<<<dim3(MTOT / 32, DIN / 64), 32, 0, stream>>>(dt_h, Wh_dt, b_dt, deltaf);

    // 7) chunked selective scan (3 passes), fused D-skip + SiLU gate -> yg f16
    scan_chunk_kernel<<<NB * (DIN / 2) * NCH, 32, 0, stream>>>(deltaf, xc_h, dtBC, A_log, Pbuf, Hbuf);
    scan_combine_kernel<<<NB * (DIN / 2), 32, 0, stream>>>(Pbuf, Hbuf, carry);
    scan_final_kernel<<<NB * (DIN / 2) * NCH, 32, 0, stream>>>(deltaf, xc_h, dtBC, z_h, A_log, Dpar, carry, yg_h);

    // 8) out_proj (WMMA) -> d_out (B, C, L)
    gemm_outproj<<<dim3(MTOT / 32, DIMC / 64), 32, 0, stream>>>(yg_h, Wh_out, out);
}
